// MSDeformAttnTransformer_31490700214566
// MI455X (gfx1250) — matrix operand order repaired and checked by reading, not compiled, not run on hardware
//
#include <hip/hip_runtime.h>
#include <hip/hip_bf16.h>

#define LEN    21760
#define MTOT   43520   // B * LEN
#define EMBED  256
#define FFN    1024
#define LAYERS 6
#define DH     32

typedef __attribute__((ext_vector_type(16))) __bf16 v16bf;
typedef __attribute__((ext_vector_type(8)))  __bf16 v8bf;
typedef __attribute__((ext_vector_type(8)))  float  v8f;
typedef int v4i_ __attribute__((vector_size(16)));   // payload type of the b128 async builtins

// ---------------------------------------------------------------------------
// CDNA5 async global->LDS copy (ASYNCcnt-tracked), with synchronous fallback
// if this toolchain does not declare the gfx1250 async builtins.
// Builtin signature (per hipcc diagnostic): (v4i AS1*, v4i AS3*, imm offset, imm cpol).
// ---------------------------------------------------------------------------
#if defined(__AMDGCN__) && __has_builtin(__builtin_amdgcn_global_load_async_to_lds_b128)
#define HAS_ASYNC_LDS 1
#else
#define HAS_ASYNC_LDS 0
#endif

__device__ __forceinline__ void copy16_to_lds(const __bf16* __restrict__ g,
                                              __bf16* __restrict__ l)
{
#if HAS_ASYNC_LDS
  __builtin_amdgcn_global_load_async_to_lds_b128(
      (__attribute__((address_space(1))) v4i_*)g,
      (__attribute__((address_space(3))) v4i_*)l, 0, 0);
#else
  *(v8bf*)l = *(const v8bf*)g;
#endif
}

__device__ __forceinline__ void async_copy_wait()
{
#if HAS_ASYNC_LDS
#if __has_builtin(__builtin_amdgcn_s_wait_asynccnt)
  __builtin_amdgcn_s_wait_asynccnt(0);
#else
  asm volatile("s_wait_asynccnt 0x0" ::: "memory");
#endif
#endif
}

// ---------------------------------------------------------------------------
// Tiled bf16 WMMA GEMM: out[M,N] = A[M,K] * B[K,N] + bias, B given as Bt[N,K].
// Block tile 256x64, K-step 32, double-buffered LDS staged with async copies.
// 8 waves -> each wave owns a 64x32 tile = 4x2 v_wmma_f32_16x16x32_bf16.
// M % 256 == 0, N % 64 == 0, K % 32 == 0 (always true for this net).
// EPI 0: outF = acc + bias (f32).  EPI 1: outB = bf16(relu(acc + bias)).
// ---------------------------------------------------------------------------
#define BM 256
#define BN 64
#define BK 32
#define LDK 40   // padded LDS row (bf16 elems), keeps 16B alignment, skews banks

template<int EPI>
__global__ __launch_bounds__(256) void gemm_bf16_kernel(
    const __bf16* __restrict__ A, const __bf16* __restrict__ Bt,
    const float* __restrict__ bias, float* __restrict__ outF,
    __bf16* __restrict__ outB, int N, int K)
{
  __shared__ __attribute__((aligned(16))) __bf16 As[2][BM][LDK];
  __shared__ __attribute__((aligned(16))) __bf16 Bs[2][BN][LDK];

  const int tid  = threadIdx.x;
  const int lane = tid & 31;
  const int wid  = tid >> 5;
  const int wm0  = (wid & 3) * 64;   // 4 waves along M, 64 rows each
  const int wn0  = (wid >> 2) * 32;  // 2 waves along N, 32 cols each
  const int r16  = lane & 15;
  const int hl   = lane >> 4;        // which half of the wave

  const size_t row0 = (size_t)blockIdx.y * BM;
  const int    col0 = blockIdx.x * BN;

  const int lrow = tid >> 2;         // 0..63
  const int lcol = (tid & 3) * 8;    // 0,8,16,24

  const __bf16* agp = A  + (row0 + lrow) * (size_t)K + lcol;
  const __bf16* bgp = Bt + (size_t)(col0 + lrow) * K + lcol;

  // Stage one 256x32 A tile + 64x32 B tile into LDS buffer `buf`.
  auto stage = [&](int buf, int k0) {
    const __bf16* ag = agp + k0;
    copy16_to_lds(ag,                        &As[buf][lrow      ][lcol]);
    copy16_to_lds(ag +  64 * (size_t)K,      &As[buf][lrow +  64][lcol]);
    copy16_to_lds(ag + 128 * (size_t)K,      &As[buf][lrow + 128][lcol]);
    copy16_to_lds(ag + 192 * (size_t)K,      &As[buf][lrow + 192][lcol]);
    copy16_to_lds(bgp + k0,                  &Bs[buf][lrow      ][lcol]);
  };

  v8f c[4][2] = {};

  stage(0, 0);
  async_copy_wait();
  __syncthreads();

  int buf = 0;
  for (int k0 = 0; k0 < K; k0 += BK) {
    if (k0 + BK < K) stage(buf ^ 1, k0 + BK);

    // A fragment (16x32 MxK): lane<16 holds row r16, K = [0..7]+[16..23];
    // lane>=16 holds K = [8..15]+[24..31]  (ISA 7.12.2 16-bit A layout)
    v16bf a[4], b[2];
    #pragma unroll
    for (int i = 0; i < 4; ++i) {
      const __bf16* ap = &As[buf][wm0 + 16*i + r16][hl * 8];
      v8bf lo = *(const v8bf*)ap;
      v8bf hi = *(const v8bf*)(ap + 16);
      #pragma unroll
      for (int e = 0; e < 8; ++e) { a[i][e] = lo[e]; a[i][e+8] = hi[e]; }
    }
    // B fragment (32x16 KxN): lane = column n, lane<16 K=[0..7]+[16..23];
    // lane>=16 K=[8..15]+[24..31] (same K order as A).
    #pragma unroll
    for (int j = 0; j < 2; ++j) {
      const __bf16* bp = &Bs[buf][wn0 + 16*j + r16][hl * 8];
      v8bf lo = *(const v8bf*)bp;
      v8bf hi = *(const v8bf*)(bp + 16);
      #pragma unroll
      for (int e = 0; e < 8; ++e) { b[j][e] = lo[e]; b[j][e+8] = hi[e]; }
    }
    #pragma unroll
    for (int i = 0; i < 4; ++i)
      #pragma unroll
      for (int j = 0; j < 2; ++j)
        c[i][j] = __builtin_amdgcn_wmma_f32_16x16x32_bf16(
            false, a[i], false, b[j], (short)0, c[i][j], false, false);

    async_copy_wait();   // my async stage of the other buffer is complete
    __syncthreads();     // whole block done computing `buf` / staging `buf^1`
    buf ^= 1;
  }

  // C/D layout: lane<16 -> N=lane, M=r; lane>=16 -> N=lane-16, M=r+8
  #pragma unroll
  for (int i = 0; i < 4; ++i) {
    #pragma unroll
    for (int j = 0; j < 2; ++j) {
      const int n = col0 + wn0 + 16*j + r16;
      const float bb = bias[n];
      const size_t mb = row0 + wm0 + 16*i + hl * 8;
      #pragma unroll
      for (int r = 0; r < 8; ++r) {
        const size_t o = (mb + r) * (size_t)N + n;
        float vv = c[i][j][r] + bb;
        if (EPI == 0) outF[o] = vv;
        else          outB[o] = (__bf16)fmaxf(vv, 0.f);
      }
    }
  }
}

// ---------------------------------------------------------------------------
// Flatten (B,C,H,W) -> x[(b,tok),c] and pos + level_embed.
// ---------------------------------------------------------------------------
__global__ void flatten_kernel(const float* __restrict__ src, const float* __restrict__ pos,
                               const float* __restrict__ le, float* __restrict__ xf,
                               float* __restrict__ posf, int start, int HW)
{
  int i = blockIdx.x * 256 + threadIdx.x;        // over B*256*HW, multiple of 256
  int w = i % HW;
  int c = (i / HW) & 255;
  int b = i / (HW * 256);
  size_t o = ((size_t)b * LEN + start + w) * 256 + c;
  xf[o]   = src[i];
  posf[o] = pos[i] + le[c];
}

// (LAYERS,K,N) f32 -> (LAYERS,N,K) bf16
__global__ void wtrans_kernel(const float* __restrict__ W, __bf16* __restrict__ Wt,
                              int K, int N)
{
  size_t i = (size_t)blockIdx.x * 256 + threadIdx.x;
  int n = (int)(i % N);
  int k = (int)((i / N) % K);
  int l = (int)(i / ((size_t)N * K));
  Wt[((size_t)l * N + n) * K + k] = (__bf16)W[i];
}

__global__ void prep_q_kernel(const float* __restrict__ x, const float* __restrict__ pos,
                              __bf16* __restrict__ qbf, __bf16* __restrict__ xbf)
{
  size_t i = (size_t)blockIdx.x * 256 + threadIdx.x;
  float xv = x[i];
  qbf[i] = (__bf16)(xv + pos[i]);
  xbf[i] = (__bf16)xv;
}

// softmax over 16 contiguous logits per (token, head), in place
__global__ void softmax16_kernel(float* __restrict__ awl)
{
  size_t i = (size_t)blockIdx.x * 256 + threadIdx.x;
  float* p = awl + i * 16;
  float mx = p[0];
  #pragma unroll
  for (int j = 1; j < 16; ++j) mx = fmaxf(mx, p[j]);
  float e[16], s = 0.f;
  #pragma unroll
  for (int j = 0; j < 16; ++j) { e[j] = __expf(p[j] - mx); s += e[j]; }
  float inv = 1.f / s;
  #pragma unroll
  for (int j = 0; j < 16; ++j) p[j] = e[j] * inv;
}

__device__ __forceinline__ float corner_tap(const float* __restrict__ vl, int W, int H,
                                            int cx, int cy, float w)
{
  bool valid = (cx >= 0) && (cx < W) && (cy >= 0) && (cy < H);
  int ix = min(max(cx, 0), W - 1);
  int iy = min(max(cy, 0), H - 1);
  float g = vl[(size_t)(iy * W + ix) * EMBED];
  return valid ? g * w : 0.f;
}

// One wave = one (token, head); lanes = Dh channels -> 128B-contiguous gathers.
__global__ void sample_kernel(const float* __restrict__ v, const float* __restrict__ off,
                              const float* __restrict__ aw, __bf16* __restrict__ accbf)
{
  const int gid = blockIdx.x * 256 + threadIdx.x;   // M*256 threads
  const int d = gid & 31;
  const int h = (gid >> 5) & 7;
  const int m = gid >> 8;
  const int b = m / LEN;
  const int t = m - b * LEN;

  const int Sz[4] = {128, 64, 32, 16};
  const int S0[4] = {0, 16384, 20480, 21504};

  int lvl_t = 3, lo = t - 21504;
  if (t < 16384)      { lvl_t = 0; lo = t; }
  else if (t < 20480) { lvl_t = 1; lo = t - 16384; }
  else if (t < 21504) { lvl_t = 2; lo = t - 20480; }
  const int Wt_ = Sz[lvl_t];
  const float refx = ((lo % Wt_) + 0.5f) / (float)Wt_;
  const float refy = ((lo / Wt_) + 0.5f) / (float)Wt_;   // square levels

  const float* offrow = off + (size_t)m * 256 + h * 32;
  const float* awrow  = aw  + (size_t)m * 128 + h * 16;
  const float* vbase  = v + (size_t)b * LEN * EMBED + h * DH + d;

  float acc = 0.f;
  #pragma unroll
  for (int lvl = 0; lvl < 4; ++lvl) {
    const int Wl = Sz[lvl], Hl = Sz[lvl];
    const float fW = (float)Wl, fH = (float)Hl;
    const float* vl = vbase + (size_t)S0[lvl] * EMBED;
    #pragma unroll
    for (int p = 0; p < 4; ++p) {
      float ox = offrow[(lvl * 4 + p) * 2 + 0];
      float oy = offrow[(lvl * 4 + p) * 2 + 1];
      float a  = awrow[lvl * 4 + p];
      float sx = (refx + ox / fW) * fW - 0.5f;
      float sy = (refy + oy / fH) * fH - 0.5f;
      float x0 = floorf(sx), y0 = floorf(sy);
      float dx = sx - x0,    dy = sy - y0;
      int xi = (int)x0, yi = (int)y0;
      float s =
          corner_tap(vl, Wl, Hl, xi,     yi,     (1.f - dx) * (1.f - dy)) +
          corner_tap(vl, Wl, Hl, xi + 1, yi,     dx * (1.f - dy)) +
          corner_tap(vl, Wl, Hl, xi,     yi + 1, (1.f - dx) * dy) +
          corner_tap(vl, Wl, Hl, xi + 1, yi + 1, dx * dy);
      acc += a * s;
    }
  }
  accbf[(size_t)gid] = (__bf16)acc;   // layout (m, h*32+d) == gid
}

// x = LN(x + attn) * g + b ; also emit bf16 copy for the FFN GEMM
__global__ void ln_kernel(const float* __restrict__ attn, const float* __restrict__ gv,
                          const float* __restrict__ bv, float* __restrict__ x,
                          __bf16* __restrict__ xbf)
{
  const int m = blockIdx.x;
  const int c = threadIdx.x;
  const size_t o = (size_t)m * 256 + c;
  const float e = x[o] + attn[o];
  __shared__ float red[256];
  red[c] = e;
  __syncthreads();
  for (int s = 128; s > 0; s >>= 1) { if (c < s) red[c] += red[c + s]; __syncthreads(); }
  const float mu = red[0] * (1.f / 256.f);
  __syncthreads();
  const float ctr = e - mu;
  red[c] = ctr * ctr;
  __syncthreads();
  for (int s = 128; s > 0; s >>= 1) { if (c < s) red[c] += red[c + s]; __syncthreads(); }
  const float var = red[0] * (1.f / 256.f);
  const float y = ctr * rsqrtf(var + 1e-5f) * gv[c] + bv[c];
  x[o]   = y;
  xbf[o] = (__bf16)y;
}

// ---------------------------------------------------------------------------
extern "C" void kernel_launch(void* const* d_in, const int* in_sizes, int n_in,
                              void* d_out, int out_size, void* d_ws, size_t ws_size,
                              hipStream_t stream)
{
  // setup_inputs order: src0,pos0,src1,pos1,src2,pos2,src3,pos3,level_embed,
  // Wv,bv,Woff,boff,Waw,baw,Wout,bout,gamma,beta,W1,b1,W2,b2
  const float* src[4]  = {(const float*)d_in[0], (const float*)d_in[2],
                          (const float*)d_in[4], (const float*)d_in[6]};
  const float* posi[4] = {(const float*)d_in[1], (const float*)d_in[3],
                          (const float*)d_in[5], (const float*)d_in[7]};
  const float* level_embed = (const float*)d_in[8];
  const float* Wv   = (const float*)d_in[9];
  const float* bv   = (const float*)d_in[10];
  const float* Woff = (const float*)d_in[11];
  const float* boff = (const float*)d_in[12];
  const float* Waw  = (const float*)d_in[13];
  const float* baw  = (const float*)d_in[14];
  const float* Wout = (const float*)d_in[15];
  const float* bout = (const float*)d_in[16];
  const float* gmm  = (const float*)d_in[17];
  const float* bta  = (const float*)d_in[18];
  const float* W1   = (const float*)d_in[19];
  const float* b1   = (const float*)d_in[20];
  const float* W2   = (const float*)d_in[21];
  const float* b2   = (const float*)d_in[22];

  size_t cur = 0;
  auto alloc = [&](size_t bytes) -> void* {
    cur = (cur + 255) & ~(size_t)255;
    void* p = (char*)d_ws + cur;
    cur += bytes;
    return p;
  };
  const size_t MC = (size_t)MTOT * EMBED;
  float*  xf    = (float*) alloc(MC * 4);
  float*  posf  = (float*) alloc(MC * 4);
  __bf16* qbf   = (__bf16*)alloc(MC * 2);                 // reused as acc_bf
  __bf16* xbf   = (__bf16*)alloc(MC * 2);
  float*  offb  = (float*) alloc(MC * 4);                 // reused as attn out
  float*  awl   = (float*) alloc((size_t)MTOT * 128 * 4); // logits, softmax in place
  float*  vf    = (float*) alloc(MC * 4);
  __bf16* hid   = (__bf16*)alloc((size_t)MTOT * FFN * 2);
  __bf16* WvT   = (__bf16*)alloc((size_t)LAYERS * 256 * 256 * 2);
  __bf16* WoffT = (__bf16*)alloc((size_t)LAYERS * 256 * 256 * 2);
  __bf16* WawT  = (__bf16*)alloc((size_t)LAYERS * 128 * 256 * 2);
  __bf16* WoutT = (__bf16*)alloc((size_t)LAYERS * 256 * 256 * 2);
  __bf16* W1T   = (__bf16*)alloc((size_t)LAYERS * 1024 * 256 * 2);
  __bf16* W2T   = (__bf16*)alloc((size_t)LAYERS * 256 * 1024 * 2);
  (void)ws_size; (void)in_sizes; (void)n_in; (void)out_size;

  // Flatten levels into (M, 256) token-major layout
  const int HWs[4] = {16384, 4096, 1024, 256};
  const int S0s[4] = {0, 16384, 20480, 21504};
  for (int l = 0; l < 4; ++l) {
    int tot = 2 * 256 * HWs[l];
    flatten_kernel<<<tot / 256, 256, 0, stream>>>(src[l], posi[l],
                                                  level_embed + l * 256,
                                                  xf, posf, S0s[l], HWs[l]);
  }
  // Weights -> bf16, transposed to (N, K)
  wtrans_kernel<<<(LAYERS*256*256 )/256, 256, 0, stream>>>(Wv,   WvT,   256, 256);
  wtrans_kernel<<<(LAYERS*256*256 )/256, 256, 0, stream>>>(Woff, WoffT, 256, 256);
  wtrans_kernel<<<(LAYERS*256*128 )/256, 256, 0, stream>>>(Waw,  WawT,  256, 128);
  wtrans_kernel<<<(LAYERS*256*256 )/256, 256, 0, stream>>>(Wout, WoutT, 256, 256);
  wtrans_kernel<<<(LAYERS*256*1024)/256, 256, 0, stream>>>(W1,   W1T,   256, 1024);
  wtrans_kernel<<<(LAYERS*1024*256)/256, 256, 0, stream>>>(W2,   W2T,   1024, 256);

  const int MB = MTOT / BM;  // 170
  for (int li = 0; li < LAYERS; ++li) {
    prep_q_kernel<<<MTOT, 256, 0, stream>>>(xf, posf, qbf, xbf);
    gemm_bf16_kernel<0><<<dim3(256/BN,  MB), 256, 0, stream>>>(
        qbf, WoffT + (size_t)li*256*256, boff + li*256, offb, nullptr, 256, 256);
    gemm_bf16_kernel<0><<<dim3(128/BN,  MB), 256, 0, stream>>>(
        qbf, WawT  + (size_t)li*128*256, baw  + li*128, awl,  nullptr, 128, 256);
    gemm_bf16_kernel<0><<<dim3(256/BN,  MB), 256, 0, stream>>>(
        xbf, WvT   + (size_t)li*256*256, bv   + li*256, vf,   nullptr, 256, 256);
    softmax16_kernel<<<(MTOT*8)/256, 256, 0, stream>>>(awl);
    sample_kernel<<<MTOT, 256, 0, stream>>>(vf, offb, awl, qbf /*acc_bf*/);
    gemm_bf16_kernel<0><<<dim3(256/BN,  MB), 256, 0, stream>>>(
        qbf, WoutT + (size_t)li*256*256, bout + li*256, offb /*attn*/, nullptr, 256, 256);
    ln_kernel<<<MTOT, 256, 0, stream>>>(offb, gmm + li*256, bta + li*256, xf, xbf);
    gemm_bf16_kernel<1><<<dim3(FFN/BN,  MB), 256, 0, stream>>>(
        xbf, W1T + (size_t)li*1024*256, b1 + li*1024, nullptr, hid, 1024, 256);
    gemm_bf16_kernel<0><<<dim3(256/BN,  MB), 256, 0, stream>>>(
        hid, W2T + (size_t)li*256*1024, b2 + li*256, xf, nullptr, 256, 1024);
  }
  (void)hipMemcpyAsync(d_out, xf, MC * 4, hipMemcpyDeviceToDevice, stream);
}
